// QuantGRU_52862457480002
// MI455X (gfx1250) — 8-bit matrix operand order repaired and checked by reading, compiled, not run on hardware
//
#include <hip/hip_runtime.h>
#include <stdint.h>


typedef int v8i __attribute__((ext_vector_type(8)));

#define T_DIM 512
#define B_DIM 64
#define I_DIM 512
#define H_DIM 512
#define H3    (3 * H_DIM)

__device__ __forceinline__ float fqs(float x, float s, float qlo, float qhi) {
    float q = fminf(fmaxf(rintf(x * s), qlo), qhi);
    return q / s;
}
__device__ __forceinline__ int8_t q8(float x, float s) {
    float q = fminf(fmaxf(rintf(x * s), -128.0f), 127.0f);
    return (int8_t)(int)q;
}

__global__ void k_quant_x(const float* __restrict__ x, int8_t* __restrict__ xq, int n) {
    int i = blockIdx.x * blockDim.x + threadIdx.x;
    if (i < n) xq[i] = q8(x[i], 16.0f);
}

__global__ void k_quant_mat_cm(const float* __restrict__ w, int8_t* __restrict__ wq, int n) {
    int i = blockIdx.x * blockDim.x + threadIdx.x;
    if (i < n) {
        int col = i >> 9, k = i & 511;
        wq[i] = q8(w[(size_t)k * H3 + col], 1024.0f);
    }
}

__global__ void k_quant_bias(const float* __restrict__ b, float* __restrict__ bq, int n) {
    int i = blockIdx.x * blockDim.x + threadIdx.x;
    if (i < n) bq[i] = fqs(b[i], 256.0f, -128.0f, 127.0f);
}

__global__ void k_quant_h0(const float* __restrict__ h0, int8_t* __restrict__ hq, int n) {
    int i = blockIdx.x * blockDim.x + threadIdx.x;
    if (i < n) hq[i] = q8(h0[i], 128.0f);
}

__global__ void k_wx_gemm(const int8_t* __restrict__ xq,
                          const int8_t* __restrict__ wq,
                          int8_t* __restrict__ wxq) {
    const int lane = threadIdx.x;
    const int hi   = lane >> 4;
    const int ln   = lane & 15;
    const int rowT = blockIdx.x * 16;
    const int colT = blockIdx.y * 16;

    const int8_t* arow = xq + (size_t)(rowT + ln) * I_DIM;
    const int8_t* bcol = wq + (size_t)(colT + ln) * I_DIM;
    const int ako = 16 * hi;
    const int bko = hi ? 16 : 0;

    v8i acc = {0, 0, 0, 0, 0, 0, 0, 0};
#pragma unroll
    for (int k0 = 0; k0 < I_DIM; k0 += 64) {
        v8i a, b;
#pragma unroll
        for (int g = 0; g < 4; ++g) {
            int2 av = *(const int2*)(arow + k0 + 32 * (g >> 1) + 8 * (g & 1) + ako);
            a[2 * g] = av.x; a[2 * g + 1] = av.y;
        }
        int4 b0 = *(const int4*)(bcol + k0 + bko);
        int4 b1 = *(const int4*)(bcol + k0 + bko + 32);
        b[0] = b0.x; b[1] = b0.y; b[2] = b0.z; b[3] = b0.w;
        b[4] = b1.x; b[5] = b1.y; b[6] = b1.z; b[7] = b1.w;
        acc = __builtin_amdgcn_wmma_i32_16x16x64_iu8(true, a, true, b, acc, false, false);
    }
#pragma unroll
    for (int i = 0; i < 8; ++i) {
        int row = rowT + i + 8 * hi;
        int col = colT + ln;
        float q = fminf(fmaxf(rintf((float)acc[i] * (1.0f / 1024.0f)), -128.0f), 127.0f);
        wxq[(size_t)row * H3 + col] = (int8_t)(int)q;
    }
}

__global__ void k_step(const int8_t* __restrict__ hq_in,
                       int8_t* __restrict__ hq_out,
                       const int8_t* __restrict__ rq,
                       const int8_t* __restrict__ wxq,
                       const float* __restrict__ bxq,
                       const float* __restrict__ brq,
                       float* __restrict__ out, int t) {
    const int lane = threadIdx.x;
    const int hi   = lane >> 4;
    const int ln   = lane & 15;
    const int mT = blockIdx.x * 16;
    const int nT = blockIdx.y * 16;

    const int8_t* arow = hq_in + (size_t)(mT + ln) * H_DIM;
    const int colZ = nT + ln;
    const int8_t* bZ = rq + (size_t)(colZ) * H_DIM;
    const int8_t* bR = rq + (size_t)(colZ + H_DIM) * H_DIM;
    const int8_t* bG = rq + (size_t)(colZ + 2 * H_DIM) * H_DIM;
    const int ako = 16 * hi;
    const int bko = hi ? 16 : 0;

    v8i accZ = {0, 0, 0, 0, 0, 0, 0, 0};
    v8i accR = {0, 0, 0, 0, 0, 0, 0, 0};
    v8i accG = {0, 0, 0, 0, 0, 0, 0, 0};
#pragma unroll
    for (int k0 = 0; k0 < H_DIM; k0 += 64) {
        v8i a, b;
#pragma unroll
        for (int g = 0; g < 4; ++g) {
            int2 av = *(const int2*)(arow + k0 + 32 * (g >> 1) + 8 * (g & 1) + ako);
            a[2 * g] = av.x; a[2 * g + 1] = av.y;
        }
        {   int4 b0 = *(const int4*)(bZ + k0 + bko);
            int4 b1 = *(const int4*)(bZ + k0 + bko + 32);
            b[0]=b0.x; b[1]=b0.y; b[2]=b0.z; b[3]=b0.w;
            b[4]=b1.x; b[5]=b1.y; b[6]=b1.z; b[7]=b1.w;
            accZ = __builtin_amdgcn_wmma_i32_16x16x64_iu8(true, a, true, b, accZ, false, false); }
        {   int4 b0 = *(const int4*)(bR + k0 + bko);
            int4 b1 = *(const int4*)(bR + k0 + bko + 32);
            b[0]=b0.x; b[1]=b0.y; b[2]=b0.z; b[3]=b0.w;
            b[4]=b1.x; b[5]=b1.y; b[6]=b1.z; b[7]=b1.w;
            accR = __builtin_amdgcn_wmma_i32_16x16x64_iu8(true, a, true, b, accR, false, false); }
        {   int4 b0 = *(const int4*)(bG + k0 + bko);
            int4 b1 = *(const int4*)(bG + k0 + bko + 32);
            b[0]=b0.x; b[1]=b0.y; b[2]=b0.z; b[3]=b0.w;
            b[4]=b1.x; b[5]=b1.y; b[6]=b1.z; b[7]=b1.w;
            accG = __builtin_amdgcn_wmma_i32_16x16x64_iu8(true, a, true, b, accG, false, false); }
    }

    const int nZ = nT + ln;
#pragma unroll
    for (int i = 0; i < 8; ++i) {
        const int mb = mT + i + 8 * hi;
        float rz = fminf(fmaxf(rintf((float)accZ[i] * (1.0f / 8192.0f)), -128.0f), 127.0f) * (1.0f / 16.0f);
        float rr = fminf(fmaxf(rintf((float)accR[i] * (1.0f / 8192.0f)), -128.0f), 127.0f) * (1.0f / 16.0f);
        float rg = fminf(fmaxf(rintf((float)accG[i] * (1.0f / 8192.0f)), -128.0f), 127.0f) * (1.0f / 16.0f);
        float rbz = fqs(rz + brq[nZ],             16.0f, -128.0f, 127.0f);
        float rbr = fqs(rr + brq[nZ + H_DIM],     16.0f, -128.0f, 127.0f);
        float rbg = fqs(rg + brq[nZ + 2 * H_DIM], 16.0f, -128.0f, 127.0f);
        const size_t wrow = ((size_t)t * B_DIM + mb) * H3;
        float wz = (float)wxq[wrow + nZ]             * (1.0f / 16.0f);
        float wr = (float)wxq[wrow + nZ + H_DIM]     * (1.0f / 16.0f);
        float wg = (float)wxq[wrow + nZ + 2 * H_DIM] * (1.0f / 16.0f);
        float pz = fqs(wz + bxq[nZ] + rbz, 8.0f, -128.0f, 127.0f);
        float z  = fqs(1.0f / (1.0f + expf(-pz)), 256.0f, 0.0f, 255.0f);
        float pr = fqs(wr + bxq[nZ + H_DIM] + rbr, 8.0f, -128.0f, 127.0f);
        float r  = fqs(1.0f / (1.0f + expf(-pr)), 256.0f, 0.0f, 255.0f);
        float rRh = fqs(r * rbg, 16.0f, -128.0f, 127.0f);
        float pg  = fqs(wg + bxq[nZ + 2 * H_DIM] + rRh, 8.0f, -128.0f, 127.0f);
        float g   = fqs(tanhf(pg), 128.0f, -128.0f, 127.0f);
        float hp  = (float)hq_in[(size_t)mb * H_DIM + nZ] * (1.0f / 128.0f);
        float oldc = fqs(z * hp, 128.0f, -128.0f, 127.0f);
        float newc = fqs((1.0f - z) * g, 128.0f, -128.0f, 127.0f);
        float hn   = fqs(oldc + newc, 128.0f, -128.0f, 127.0f);
        out[((size_t)t * B_DIM + mb) * H_DIM + nZ] = hn;
        hq_out[(size_t)mb * H_DIM + nZ] = (int8_t)(int)rintf(hn * 128.0f);
    }
}

extern "C" void kernel_launch(void* const* d_in, const int* in_sizes, int n_in,
                              void* d_out, int out_size, void* d_ws, size_t ws_size,
                              hipStream_t stream) {
    const float* x  = (const float*)d_in[0];
    const float* h0 = (const float*)d_in[1];
    const float* W  = (const float*)d_in[2];
    const float* R  = (const float*)d_in[3];
    const float* bx = (const float*)d_in[4];
    const float* br = (const float*)d_in[5];
    float* out = (float*)d_out;

    uint8_t* ws = (uint8_t*)d_ws;
    size_t off = 0;
    auto carve = [&](size_t bytes) -> uint8_t* {
        uint8_t* p = ws + off;
        off = (off + bytes + 255) & ~(size_t)255;
        return p;
    };
    int8_t* xq   = (int8_t*)carve((size_t)T_DIM * B_DIM * I_DIM);
    int8_t* wq   = (int8_t*)carve((size_t)I_DIM * H3);
    int8_t* rq   = (int8_t*)carve((size_t)H_DIM * H3);
    int8_t* wxq  = (int8_t*)carve((size_t)T_DIM * B_DIM * H3);
    float*  bxq  = (float*)carve(sizeof(float) * H3);
    float*  brq  = (float*)carve(sizeof(float) * H3);
    int8_t* hbuf0 = (int8_t*)carve((size_t)B_DIM * H_DIM);
    int8_t* hbuf1 = (int8_t*)carve((size_t)B_DIM * H_DIM);
    (void)ws_size; (void)n_in; (void)in_sizes; (void)out_size;

    {
        int n = T_DIM * B_DIM * I_DIM;
        k_quant_x<<<(n + 255) / 256, 256, 0, stream>>>(x, xq, n);
    }
    {
        int n = I_DIM * H3;
        k_quant_mat_cm<<<(n + 255) / 256, 256, 0, stream>>>(W, wq, n);
        k_quant_mat_cm<<<(n + 255) / 256, 256, 0, stream>>>(R, rq, n);
    }
    k_quant_bias<<<(H3 + 255) / 256, 256, 0, stream>>>(bx, bxq, H3);
    k_quant_bias<<<(H3 + 255) / 256, 256, 0, stream>>>(br, brq, H3);
    {
        int n = B_DIM * H_DIM;
        k_quant_h0<<<(n + 255) / 256, 256, 0, stream>>>(h0, hbuf0, n);
    }

    k_wx_gemm<<<dim3((T_DIM * B_DIM) / 16, H3 / 16), 32, 0, stream>>>(xq, wq, wxq);

    for (int t = 0; t < T_DIM; ++t) {
        const int8_t* hin = (t & 1) ? hbuf1 : hbuf0;
        int8_t*       hout = (t & 1) ? hbuf0 : hbuf1;
        k_step<<<dim3(B_DIM / 16, H_DIM / 16), 32, 0, stream>>>(
            hin, hout, rq, wxq, bxq, brq, out, t);
    }
}
